// TransBlockV3BodyOrderTest_42305427866212
// MI455X (gfx1250) — matrix operand order repaired and checked by reading, not compiled, not run on hardware
//
#include <hip/hip_runtime.h>
#include <hip/hip_bf16.h>

// ---------------------------------------------------------------------------
// TransBlockV3 body: norm -> SO3Linear1 -> Gaunt depthwise TP -> SO3Linear2
// N=2048 nodes, L2=16 (lmax=3), C_IN=384, C_HID=256, fp32 in/out.
// Memory-bound (~100MB HBM vs ~14 GFLOP): f16 staging + v_wmma_f32_16x16x32_f16.
// Weights are pre-packed once into WMMA B-fragment order (f16) so the GEMM
// inner loop does pure vector loads + wmma.
// ---------------------------------------------------------------------------

typedef _Float16 v8h  __attribute__((ext_vector_type(8)));
typedef _Float16 v16h __attribute__((ext_vector_type(16)));
typedef float    v8f  __attribute__((ext_vector_type(8)));

#define NNODES 2048
#define CIN    384
#define CHID   256
#define L2DIM  16
#define KS1    (CIN / 32)    // 12 k-steps, gemm1
#define NT1    (CHID / 16)   // 16 col-tiles, gemm1
#define KS2    (CHID / 32)   // 8 k-steps, gemm2
#define NT2    (CIN / 16)    // 24 col-tiles, gemm2

__device__ __host__ inline int lof(int i) { return (i == 0) ? 0 : (i < 4) ? 1 : (i < 9) ? 2 : 3; }

static __device__ inline v8f wmma_f16(v16h a, v16h b, v8f c) {
  return __builtin_amdgcn_wmma_f32_16x16x32_f16(false, a, false, b, (short)0, c, false, false);
}

// ---------------------------------------------------------------------------
// K0: build Gaunt sparse table on device (Gauss-Legendre x uniform phi),
// exactly mirroring the reference quadrature. One workgroup, runs each call.
// ---------------------------------------------------------------------------
__global__ __launch_bounds__(256) void gaunt_init_kernel(int* __restrict__ koff_g,
                                                         unsigned* __restrict__ idx_g,
                                                         float* __restrict__ val_g) {
  __shared__ double xg[16], wq[16];
  __shared__ float  P4[4][4][16];
  __shared__ float  Ysh[16][512];
  __shared__ float  WQ[512];
  __shared__ float  Gs[4096];
  __shared__ int    fanin[4];
  const int    tid = threadIdx.x;
  const double PI  = 3.14159265358979323846;

  if (tid < 16) {
    // Newton iteration for roots of P_16 (Gauss-Legendre nodes, nt = 16)
    double x = cos(PI * (tid + 0.75) / 16.5);
    double dp = 1.0;
    for (int it = 0; it < 64; ++it) {
      double p0 = 1.0, p1 = x;
      for (int k = 2; k <= 16; ++k) { double pk = ((2.0*k - 1.0)*x*p1 - (k - 1.0)*p0) / k; p0 = p1; p1 = pk; }
      dp = 16.0 * (x*p1 - p0) / (x*x - 1.0);
      x -= p1 / dp;
    }
    { double p0 = 1.0, p1 = x;
      for (int k = 2; k <= 16; ++k) { double pk = ((2.0*k - 1.0)*x*p1 - (k - 1.0)*p0) / k; p0 = p1; p1 = pk; }
      dp = 16.0 * (x*p1 - p0) / (x*x - 1.0); }
    xg[tid] = x;
    wq[tid] = 2.0 / ((1.0 - x*x) * dp * dp);
  }
  __syncthreads();

  if (tid < 16) {
    // fully-normalized associated Legendre Pbar[l][m] at this theta node
    double ct = xg[tid], st = sqrt(1.0 - ct*ct);
    double P[4][4];
    for (int a = 0; a < 4; ++a) for (int b = 0; b < 4; ++b) P[a][b] = 0.0;
    P[0][0] = sqrt(1.0 / (4.0 * PI));
    for (int m = 1; m <= 3; ++m) P[m][m] = sqrt((2.0*m + 1.0) / (2.0*m)) * st * P[m-1][m-1];
    for (int m = 0; m <= 3; ++m) {
      if (m + 1 <= 3) P[m+1][m] = sqrt(2.0*m + 3.0) * ct * P[m][m];
      for (int l = m + 2; l <= 3; ++l) {
        double a = sqrt((4.0*l*l - 1.0) / ((double)l*l - (double)m*m));
        double b = sqrt((((double)l - 1.0)*((double)l - 1.0) - (double)m*m) /
                        (4.0*((double)l - 1.0)*((double)l - 1.0) - 1.0));
        P[l][m] = a * (ct * P[l-1][m] - b * P[l-2][m]);
      }
    }
    for (int l = 0; l <= 3; ++l)
      for (int m = 0; m <= 3; ++m) P4[l][m][tid] = (float)P[l][m];
  }
  __syncthreads();

  // real spherical harmonics sampled on the 16 x 32 quadrature grid
  for (int e = tid; e < 16 * 512; e += 256) {
    int idx = e >> 9, g = e & 511, t = g >> 5, p = g & 31;
    int l = lof(idx), m = idx - l*l - l;
    double phi = 2.0 * PI * (double)p / 32.0;
    float v;
    if (m == 0)      v = P4[l][0][t];
    else if (m > 0)  v = (float)(sqrt(2.0) * (double)P4[l][m][t]  * cos((double)m * phi));
    else             v = (float)(sqrt(2.0) * (double)P4[l][-m][t] * sin((double)(-m) * phi));
    Ysh[idx][g] = v;
  }
  for (int g = tid; g < 512; g += 256) WQ[g] = (float)(wq[g >> 5] * (2.0 * PI / 32.0));
  __syncthreads();

  // G[i,j,k] = sum Y_i Y_j Y_k W  (double accumulation), threshold 1e-10
  for (int e = tid; e < 4096; e += 256) {
    int i = e >> 8, j = (e >> 4) & 15, k = e & 15;
    double s = 0.0;
    for (int g = 0; g < 512; ++g)
      s += (double)Ysh[i][g] * (double)Ysh[j][g] * (double)Ysh[k][g] * (double)WQ[g];
    Gs[e] = (fabs(s) < 1e-10) ? 0.0f : (float)s;
  }
  __syncthreads();

  if (tid == 0) {
    // fan-in per l3 (count of nonzero (l1,l2) blocks)
    for (int l3 = 0; l3 < 4; ++l3) {
      int c = 0;
      for (int l1 = 0; l1 < 4; ++l1)
        for (int l2 = 0; l2 < 4; ++l2) {
          float mx = 0.f;
          for (int i = l1*l1; i < (l1+1)*(l1+1); ++i)
            for (int j = l2*l2; j < (l2+1)*(l2+1); ++j)
              for (int k = l3*l3; k < (l3+1)*(l3+1); ++k) {
                float av = fabsf(Gs[(i*16 + j)*16 + k]);
                if (av > mx) mx = av;
              }
          if (mx > 1e-8f) ++c;
        }
      fanin[l3] = (c > 0) ? c : 1;
    }
    // k-sorted sparse list: C[i,j,k]*sqrt(2l3+1)/sqrt(fanin), with packed
    // w_gaunt index (l1*4+l2)*4+l3
    int cnt = 0;
    for (int k = 0; k < 16; ++k) {
      koff_g[k] = cnt;
      int l3 = lof(k);
      float sc = sqrtf(2.f*l3 + 1.f) / sqrtf((float)fanin[l3]);
      for (int i = 0; i < 16; ++i)
        for (int j = 0; j < 16; ++j) {
          float v = Gs[(i*16 + j)*16 + k];
          if (v != 0.f) {
            idx_g[cnt] = (unsigned)(i | (j << 4) | (((lof(i)*4 + lof(j))*4 + l3) << 8));
            val_g[cnt] = v * sc;
            ++cnt;
          }
        }
    }
    koff_g[16] = cnt;
  }
}

// ---------------------------------------------------------------------------
// K0b: pre-pack per-degree fp32 weights into f16 WMMA B-fragment order:
// frag index t = ((deg*KS + ks)*NT + nt)*32 + lane, 16 contiguous halves each.
// Contiguous-k lane layout (same as A): lane<16 holds K=ks*32+p, lane>=16 K=ks*32+16+p,
// col = nt*16 + (lane&15).
// ---------------------------------------------------------------------------
__global__ __launch_bounds__(256) void pack_w_kernel(const float* __restrict__ W,
                                                     _Float16* __restrict__ out,
                                                     int Kdim, int Cout) {
  const int KS = Kdim >> 5, NT = Cout >> 4;
  const int total = 4 * KS * NT * 32;
  int t = blockIdx.x * 256 + threadIdx.x;
  if (t >= total) return;
  int lane = t & 31;
  int rest = t >> 5;
  int nt = rest % NT; rest /= NT;
  int ks = rest % KS;
  int deg = rest / KS;
  int col = nt * 16 + (lane & 15);
  int k0  = ks * 32 + (lane >> 4) * 16;
  const float* wp = W + ((size_t)deg * Kdim + k0) * Cout + col;
  v16h b;
#pragma unroll
  for (int p = 0; p < 16; ++p) b[p] = (_Float16)wp[p * Cout];
  *(v16h*)(out + (size_t)t * 16) = b;
}

// ---------------------------------------------------------------------------
// K1: equivariant merge-norm. One block per node; x staged into LDS with
// ASYNCcnt-tracked global->LDS async copies; writes f16 activations.
// ---------------------------------------------------------------------------
__global__ __launch_bounds__(256) void norm_kernel(const float* __restrict__ xin,
                                                   const float* __restrict__ norm_w,
                                                   const float* __restrict__ norm_b,
                                                   _Float16* __restrict__ ybuf) {
  __shared__ float xbuf[L2DIM * CIN];
  __shared__ float red[256];
  const int tid = threadIdx.x;
  const int n   = blockIdx.x;
  const float* x = xin + (size_t)n * (L2DIM * CIN);

  // async stage: 16B per lane per sweep; LDS offset = low 32 bits of the
  // generic shared-pointer (LDS aperture: addr[31:0] is the LDS byte offset)
  for (int idx = tid * 4; idx < L2DIM * CIN; idx += 256 * 4) {
    unsigned     dst = (unsigned)(unsigned long long)(&xbuf[idx]);
    const float* src = x + idx;
    asm volatile("global_load_async_to_lds_b128 %0, %1, off"
                 :: "v"(dst), "v"(src) : "memory");
  }
  asm volatile("s_wait_asynccnt 0x0" ::: "memory");
  __syncthreads();

  // mean over l=0 channels
  float s = 0.f;
  for (int c = tid; c < CIN; c += 256) s += xbuf[c];
  red[tid] = s; __syncthreads();
  for (int off = 128; off > 0; off >>= 1) { if (tid < off) red[tid] += red[tid + off]; __syncthreads(); }
  float mean = red[0] / (float)CIN;
  __syncthreads();

  // RMS over all (lm, c) with l=0 centered
  float s2 = 0.f;
  for (int idx = tid; idx < L2DIM * CIN; idx += 256) {
    float v = xbuf[idx];
    if (idx < CIN) v -= mean;
    s2 += v * v;
  }
  red[tid] = s2; __syncthreads();
  for (int off = 128; off > 0; off >>= 1) { if (tid < off) red[tid] += red[tid + off]; __syncthreads(); }
  float r = rsqrtf(red[0] / (float)(L2DIM * CIN) + 1e-5f);
  __syncthreads();

  for (int idx = tid; idx < L2DIM * CIN; idx += 256) {
    int i = idx / CIN, c = idx - i * CIN;
    float v = xbuf[idx];
    if (i == 0) v -= mean;
    v *= r * norm_w[lof(i) * CIN + c];
    if (i == 0) v += norm_b[c];
    ybuf[(size_t)n * (L2DIM * CIN) + idx] = (_Float16)v;
  }
}

// ---------------------------------------------------------------------------
// Per-degree row mapping. Rows of degree l: r -> (node=r/(2l+1),
// lm = l*l + r%(2l+1)). Row counts N*(2l+1) are all multiples of 32.
// Block = 32 rows (2 WMMA M-tiles); blockIdx.x prefix: {64,192,320,448} tiles.
// ---------------------------------------------------------------------------
__device__ inline void decode_tile(int bt, int& deg, int& r0) {
  if (bt < 64)       { deg = 0; r0 = bt * 32; }
  else if (bt < 256) { deg = 1; r0 = (bt - 64) * 32; }
  else if (bt < 576) { deg = 2; r0 = (bt - 256) * 32; }
  else               { deg = 3; r0 = (bt - 576) * 32; }
}

__device__ inline v16h load_afrag(const _Float16* __restrict__ row, int kb, int hs) {
  const _Float16* p = row + kb + hs * 16;
  v8h lo = *(const v8h*)(p);
  v8h hi = *(const v8h*)(p + 8);
  return __builtin_shufflevector(lo, hi, 0,1,2,3,4,5,6,7,8,9,10,11,12,13,14,15);
}

// ---------------------------------------------------------------------------
// K2: SO3Linear1, y_f16[N,16,384] x packed-w1 -> h_f16[N,16,256] (+b1 on l=0).
// 4 waves/block, each wave owns 64 output columns. B fragments are single
// 32-byte vector loads from the pre-packed f16 weights.
// ---------------------------------------------------------------------------
__global__ __launch_bounds__(128) void gemm1_kernel(const _Float16* __restrict__ ybuf,
                                                    const _Float16* __restrict__ wpack,
                                                    const float* __restrict__ b1,
                                                    _Float16* __restrict__ hbuf) {
  const int lane = threadIdx.x & 31;
  const int wave = threadIdx.x >> 5;
  int deg, r0; decode_tile(blockIdx.x, deg, r0);
  const int w = 2 * deg + 1;
  const int colbase = wave * 64;
  const int cb16 = colbase >> 4;
  const int rl = lane & 15, hs = lane >> 4;

  const _Float16* arow[2];
#pragma unroll
  for (int mt = 0; mt < 2; ++mt) {
    int r = r0 + mt * 16 + rl;
    int node = r / w, m = r - node * w;
    arow[mt] = ybuf + ((size_t)node * L2DIM + deg*deg + m) * CIN;
  }

  const v8f zero = {0.f,0.f,0.f,0.f,0.f,0.f,0.f,0.f};
  v8f acc[2][4];
#pragma unroll
  for (int mt = 0; mt < 2; ++mt)
#pragma unroll
    for (int nt = 0; nt < 4; ++nt) acc[mt][nt] = zero;

  for (int ks = 0; ks < KS1; ++ks) {
    v16h a0 = load_afrag(arow[0], ks * 32, hs);
    v16h a1 = load_afrag(arow[1], ks * 32, hs);
    const _Float16* bks = wpack + ((((size_t)deg * KS1 + ks) * NT1 + cb16) * 32 + lane) * 16;
#pragma unroll
    for (int nt = 0; nt < 4; ++nt) {
      v16h b = *(const v16h*)(bks + (size_t)nt * 512);
      acc[0][nt] = wmma_f16(a0, b, acc[0][nt]);
      acc[1][nt] = wmma_f16(a1, b, acc[1][nt]);
    }
  }

#pragma unroll
  for (int mt = 0; mt < 2; ++mt)
#pragma unroll
    for (int nt = 0; nt < 4; ++nt) {
      int col = colbase + nt * 16 + rl;
      float bias = (deg == 0) ? b1[col] : 0.f;  // degree 0 rows are all lm==0
#pragma unroll
      for (int j = 0; j < 8; ++j) {
        int r = r0 + mt * 16 + j + 8 * hs;
        int node = r / w, m = r - node * w;
        hbuf[((size_t)node * L2DIM + deg*deg + m) * CHID + col] =
            (_Float16)(acc[mt][nt][j] + bias);
      }
    }
}

// ---------------------------------------------------------------------------
// K3: Gaunt depthwise tensor product. block = node, thread = channel.
// t[n,k,h] = sum_{(i,j,k) in C} C_ijk * h[n,i,h] * h[n,j,h] * wg[l1,l2,l3,h]
// ---------------------------------------------------------------------------
__global__ __launch_bounds__(256) void gaunt_tp_kernel(const _Float16* __restrict__ hbuf,
                                                       const float* __restrict__ wg,
                                                       const int* __restrict__ koff_g,
                                                       const unsigned* __restrict__ idx_g,
                                                       const float* __restrict__ val_g,
                                                       _Float16* __restrict__ tbuf) {
  __shared__ float    hv[L2DIM * CHID];
  __shared__ unsigned eidx[2048];
  __shared__ float    eval[2048];
  __shared__ int      koff[17];
  const int h = threadIdx.x;
  const int n = blockIdx.x;

  if (h < 17) koff[h] = koff_g[h];
  __syncthreads();
  int cnt = koff[16]; if (cnt > 2048) cnt = 2048;
  for (int e = h; e < cnt; e += 256) { eidx[e] = idx_g[e]; eval[e] = val_g[e]; }
#pragma unroll
  for (int i = 0; i < L2DIM; ++i)
    hv[i * CHID + h] = (float)hbuf[((size_t)n * L2DIM + i) * CHID + h];
  __syncthreads();

  for (int k = 0; k < L2DIM; ++k) {
    float acc = 0.f;
    int e1 = koff[k + 1]; if (e1 > cnt) e1 = cnt;
    for (int e = koff[k]; e < e1; ++e) {
      unsigned pk = eidx[e];
      int i = pk & 15, j = (pk >> 4) & 15, widx = (pk >> 8) & 63;
      acc += eval[e] * hv[i * CHID + h] * hv[j * CHID + h] * wg[widx * CHID + h];
    }
    tbuf[((size_t)n * L2DIM + k) * CHID + h] = (_Float16)acc;
  }
}

// ---------------------------------------------------------------------------
// K4: SO3Linear2 + bias(l=0) + residual, fp32 out. 6 waves x 64 cols = 384.
// ---------------------------------------------------------------------------
__global__ __launch_bounds__(192) void gemm2_kernel(const _Float16* __restrict__ tbuf,
                                                    const _Float16* __restrict__ wpack,
                                                    const float* __restrict__ b2,
                                                    const float* __restrict__ xin,
                                                    float* __restrict__ out) {
  const int lane = threadIdx.x & 31;
  const int wave = threadIdx.x >> 5;
  int deg, r0; decode_tile(blockIdx.x, deg, r0);
  const int w = 2 * deg + 1;
  const int colbase = wave * 64;
  const int cb16 = colbase >> 4;
  const int rl = lane & 15, hs = lane >> 4;

  const _Float16* arow[2];
#pragma unroll
  for (int mt = 0; mt < 2; ++mt) {
    int r = r0 + mt * 16 + rl;
    int node = r / w, m = r - node * w;
    arow[mt] = tbuf + ((size_t)node * L2DIM + deg*deg + m) * CHID;
  }

  const v8f zero = {0.f,0.f,0.f,0.f,0.f,0.f,0.f,0.f};
  v8f acc[2][4];
#pragma unroll
  for (int mt = 0; mt < 2; ++mt)
#pragma unroll
    for (int nt = 0; nt < 4; ++nt) acc[mt][nt] = zero;

  for (int ks = 0; ks < KS2; ++ks) {
    v16h a0 = load_afrag(arow[0], ks * 32, hs);
    v16h a1 = load_afrag(arow[1], ks * 32, hs);
    const _Float16* bks = wpack + ((((size_t)deg * KS2 + ks) * NT2 + cb16) * 32 + lane) * 16;
#pragma unroll
    for (int nt = 0; nt < 4; ++nt) {
      v16h b = *(const v16h*)(bks + (size_t)nt * 512);
      acc[0][nt] = wmma_f16(a0, b, acc[0][nt]);
      acc[1][nt] = wmma_f16(a1, b, acc[1][nt]);
    }
  }

#pragma unroll
  for (int mt = 0; mt < 2; ++mt)
#pragma unroll
    for (int nt = 0; nt < 4; ++nt) {
      int col = colbase + nt * 16 + rl;
      float bias = (deg == 0) ? b2[col] : 0.f;
#pragma unroll
      for (int j = 0; j < 8; ++j) {
        int r = r0 + mt * 16 + j + 8 * hs;
        int node = r / w, m = r - node * w;
        size_t o = ((size_t)node * L2DIM + deg*deg + m) * CIN + col;
        out[o] = acc[mt][nt][j] + bias + xin[o];
      }
    }
}

// ---------------------------------------------------------------------------
// Host launch. Inputs (setup_inputs order): inputs, batch, norm_w, norm_b,
// w1, b1, w_gaunt, w2, b2. batch is unused (dropout disabled).
// ---------------------------------------------------------------------------
extern "C" void kernel_launch(void* const* d_in, const int* in_sizes, int n_in,
                              void* d_out, int out_size, void* d_ws, size_t ws_size,
                              hipStream_t stream) {
  const float* inputs = (const float*)d_in[0];
  const float* norm_w = (const float*)d_in[2];
  const float* norm_b = (const float*)d_in[3];
  const float* w1     = (const float*)d_in[4];
  const float* b1     = (const float*)d_in[5];
  const float* wg     = (const float*)d_in[6];
  const float* w2     = (const float*)d_in[7];
  const float* b2     = (const float*)d_in[8];
  float* out = (float*)d_out;

  char* ws = (char*)d_ws;
  int*      koff  = (int*)(ws);
  unsigned* idxl  = (unsigned*)(ws + 128);
  float*    vall  = (float*)(ws + 128 + 4096 * 4);
  _Float16* wp1   = (_Float16*)(ws + 65536);                       // 786432 B
  _Float16* wp2   = (_Float16*)(ws + 65536 + 786432);              // 786432 B
  _Float16* ybuf  = (_Float16*)(ws + 65536 + 2 * 786432);          // 25165824 B
  _Float16* hbuf  = (_Float16*)(ws + 65536 + 2 * 786432 + (size_t)NNODES * L2DIM * CIN * 2);
  _Float16* tbuf  = ybuf;  // y dead after K2; reuse region for t

  gaunt_init_kernel<<<1, 256, 0, stream>>>(koff, idxl, vall);
  pack_w_kernel<<<96, 256, 0, stream>>>(w1, wp1, CIN, CHID);   // 4*12*16*32 = 24576 frags
  pack_w_kernel<<<96, 256, 0, stream>>>(w2, wp2, CHID, CIN);   // 4*8*24*32  = 24576 frags
  norm_kernel<<<NNODES, 256, 0, stream>>>(inputs, norm_w, norm_b, ybuf);
  gemm1_kernel<<<1024, 128, 0, stream>>>(ybuf, wp1, b1, hbuf);
  gaunt_tp_kernel<<<NNODES, 256, 0, stream>>>(hbuf, wg, koff, idxl, vall, tbuf);
  gemm2_kernel<<<1024, 192, 0, stream>>>(tbuf, wp2, b2, inputs, out);
}
